// EigCoeffReg_6906307412540
// MI455X (gfx1250) — matrix operand order repaired and checked by reading, not compiled, not run on hardware
//
#include <hip/hip_runtime.h>
#include <hip/hip_bf16.h>
#include <math.h>

// ---------------------------------------------------------------------------
// Problem constants (match the reference)
// ---------------------------------------------------------------------------
#define DDIM 1024      // latent dim D
#define NSAMP 8192     // zk rows B
#define QR_ITERS 12    // QR-algorithm sweeps (approximate Schur form)
#define SCALE 0.1f

typedef _Float16 f16;
typedef __attribute__((ext_vector_type(16))) _Float16 v16h;
typedef __attribute__((ext_vector_type(8)))  float    v8f;
typedef __attribute__((ext_vector_type(8)))  unsigned int v8u;
typedef __attribute__((ext_vector_type(4)))  unsigned int u32x4;
typedef __attribute__((ext_vector_type(8)))  int i32x8;
typedef __attribute__((ext_vector_type(4)))  int i32x4;

union FragA { v8u u; v16h h; };

// ---------------------------------------------------------------------------
// Shared WMMA tiling parameters
//   block = 256 threads = 8 waves; tile BM=128 x BN=64
//   wave w -> rows [16w,16w+16); 4 N-fragments of 16 cols each
//   fragment gathers follow the ISA VGPR layouts:
//     A frag (lane L: r=L&15,h=L>>4; vgpr v): Kpair = (v>>2)*8 + h*4 + (v&3)
//     B frag (lane L: n=L&15,h=L>>4; vgpr v): Kpair = (v>>2)*8 + h*4 + (v&3)
// ---------------------------------------------------------------------------
#define BM 128
#define BN 64

// ======================= f32-input GEMM (QR sweeps) ========================
#define KB 32
#define KH (KB/2)      // u32 (f16-pair) entries per row
#define LDP (KH+1)     // padded LDS row stride

__device__ __forceinline__ unsigned int pack_h2(float x, float y) {
    union { unsigned int u; f16 h[2]; } p;
    p.h[0] = (f16)x; p.h[1] = (f16)y;
    return p.u;
}

__device__ __forceinline__ void gemm_mainloop(
    const float* __restrict__ A, const float* __restrict__ Bm,
    int M, int N, int K, int m0, int n0,
    unsigned int (*As)[LDP], unsigned int (*Bs)[LDP], v8f acc[4])
{
    const int tid  = threadIdx.x;
    const int wid  = tid >> 5;
    const int lane = tid & 31;
    const int r    = lane & 15;
    const int h    = lane >> 4;
    const int mw   = wid << 4;

    for (int k0 = 0; k0 < K; k0 += KB) {
        __syncthreads();
        #pragma unroll
        for (int i = tid; i < BM * KH; i += 256) {
            int m  = i >> 4;
            int kk = i & (KH - 1);
            const float2* arow =
                reinterpret_cast<const float2*>(A + (size_t)(m0 + m) * K + k0);
            float2 v = arow[kk];
            As[m][kk] = pack_h2(v.x, v.y);
        }
        #pragma unroll
        for (int i = tid; i < KH * BN; i += 256) {
            int kk = i >> 6;
            int n  = i & (BN - 1);
            float x = Bm[(size_t)(k0 + 2 * kk)     * N + n0 + n];
            float y = Bm[(size_t)(k0 + 2 * kk + 1) * N + n0 + n];
            Bs[n][kk] = pack_h2(x, y);
        }
        if (k0 + KB < K) {
            __builtin_prefetch(A + (size_t)(m0 + (tid >> 1)) * K + k0 + KB, 0, 1);
            __builtin_prefetch(Bm + (size_t)(k0 + KB + (tid >> 6)) * N + n0 + (tid & 63), 0, 1);
        }
        __syncthreads();

        FragA fa;
        #pragma unroll
        for (int v = 0; v < 8; ++v) {
            int kk = ((v >> 2) << 3) + (h << 2) + (v & 3);
            fa.u[v] = As[mw + r][kk];
        }
        #pragma unroll
        for (int nf = 0; nf < 4; ++nf) {
            FragA fb;
            #pragma unroll
            for (int v = 0; v < 8; ++v)
                fb.u[v] = Bs[nf * 16 + r][((v >> 2) << 3) + (h << 2) + (v & 3)];
            acc[nf] = __builtin_amdgcn_wmma_f32_16x16x32_f16(
                false, fa.h, false, fb.h, (short)0, acc[nf], false, false);
        }
    }
}

__global__ __launch_bounds__(256) void gemm_wmma_kernel(
    const float* __restrict__ A, const float* __restrict__ Bm,
    float* __restrict__ C, int M, int N, int K)
{
    __shared__ unsigned int As[BM][LDP];
    __shared__ unsigned int Bs[BN][LDP];
    const int m0 = blockIdx.y * BM;
    const int n0 = blockIdx.x * BN;
    const int lane = threadIdx.x & 31;
    const int wid  = threadIdx.x >> 5;
    const int h = lane >> 4;

    v8f acc[4] = { {0,0,0,0,0,0,0,0}, {0,0,0,0,0,0,0,0},
                   {0,0,0,0,0,0,0,0}, {0,0,0,0,0,0,0,0} };
    gemm_mainloop(A, Bm, M, N, K, m0, n0, As, Bs, acc);

    #pragma unroll
    for (int nf = 0; nf < 4; ++nf) {
        int col = n0 + nf * 16 + (lane & 15);
        #pragma unroll
        for (int v = 0; v < 8; ++v) {
            int row = m0 + (wid << 4) + v + (h << 3);
            C[(size_t)row * N + col] = acc[nf][v];
        }
    }
}

// ================= TDM-staged f16 GEMM + column-sum-of-squares ==============
// S[j] += sum_i (A x B^T_t)[i,j]^2 with A: M x K f16 row-major,
// Bt: N x K f16 row-major (i.e. operand B pre-transposed so both tiles are
// contiguous along K).  Tiles staged by TENSOR_LOAD_TO_LDS, double-buffered.
#define KB2 64
#define KH2 (KB2/2)        // 32 u32 per LDS row
#define LDP2 (KH2+1)       // TDM pad: interval 32 DWORDs (code 4) + 1 DWORD

__device__ __forceinline__ void tdm_load_tile_2d(
    unsigned lds_off, const void* gaddr,
    unsigned tile_d0, unsigned tile_d1, unsigned stride_elems)
{
    unsigned long long ga = (unsigned long long)(uintptr_t)gaddr;
    u32x4 g0;
    g0[0] = 1u;                                           // count=1 (valid), user mode
    g0[1] = lds_off;                                      // LDS byte address
    g0[2] = (unsigned)(ga & 0xFFFFFFFFu);                 // global_addr[31:0]
    g0[3] = (unsigned)((ga >> 32) & 0x01FFFFFFu)          // global_addr[56:32]
          | (2u << 30);                                   // type = 2 (image)
    const unsigned td0 = 1u << 20, td1 = 1u << 20;        // generous OOB dims
    i32x8 g1;
    g1[0] = (int)((1u << 16)                              // data_size = 2 bytes
                | (1u << 20)                              // pad_enable
                | (4u << 22)                              // pad_interval = 32 DWORDs
                | (0u << 25));                            // pad_amount = 1 DWORD
    g1[1] = (int)((td0 & 0xFFFFu) << 16);                 // barrier addr=0 | td0 lo
    g1[2] = (int)(((td0 >> 16) & 0xFFFFu) | ((td1 & 0xFFFFu) << 16));
    g1[3] = (int)(((td1 >> 16) & 0xFFFFu) | (tile_d0 << 16));
    g1[4] = (int)tile_d1;                                 // tile_dim2 = 0 (2D)
    g1[5] = (int)stride_elems;                            // tensor_dim0_stride lo
    g1[6] = 0;                                            // stride hi | dim1 stride lo
    g1[7] = 0;
    i32x4 z4 = {0, 0, 0, 0};                              // groups 2/3 unused (2D)
    i32x8 z8 = {0, 0, 0, 0, 0, 0, 0, 0};                  // extra group (clang-23 form)
    __builtin_amdgcn_tensor_load_to_lds(g0, g1, z4, z4, z8, 0);
}

__global__ __launch_bounds__(256) void gemm16_colsumsq_tdm_kernel(
    const f16* __restrict__ Ah, const f16* __restrict__ Bt,
    float* __restrict__ S, int M, int N, int K)
{
    __shared__ unsigned int As[2][BM][LDP2];
    __shared__ unsigned int Bs[2][BN][LDP2];
    __shared__ float red[BN];
    const int tid  = threadIdx.x;
    const int wid  = tid >> 5;
    const int lane = tid & 31;
    const int r    = lane & 15;
    const int h    = lane >> 4;
    const int mw   = wid << 4;
    const int m0 = blockIdx.y * BM;
    const int n0 = blockIdx.x * BN;

    for (int i = tid; i < BN; i += 256) red[i] = 0.0f;

    v8f acc[4] = { {0,0,0,0,0,0,0,0}, {0,0,0,0,0,0,0,0},
                   {0,0,0,0,0,0,0,0}, {0,0,0,0,0,0,0,0} };

    // issue tile 0 into buffer 0 (single wave; TDM ignores EXEC, wave-level op)
    if (wid == 0) {
        tdm_load_tile_2d((unsigned)(uintptr_t)&As[0][0][0],
                         Ah + (size_t)m0 * K, KB2, BM, (unsigned)K);
        tdm_load_tile_2d((unsigned)(uintptr_t)&Bs[0][0][0],
                         Bt + (size_t)n0 * K, KB2, BN, (unsigned)K);
        __builtin_amdgcn_s_wait_tensorcnt(0);
    }
    __syncthreads();

    int buf = 0;
    for (int k0 = 0; k0 < K; k0 += KB2) {
        // prefetch next tile into the other buffer while computing this one
        const bool more = (k0 + KB2 < K);
        if (more && wid == 0) {
            tdm_load_tile_2d((unsigned)(uintptr_t)&As[buf ^ 1][0][0],
                             Ah + (size_t)m0 * K + (k0 + KB2), KB2, BM, (unsigned)K);
            tdm_load_tile_2d((unsigned)(uintptr_t)&Bs[buf ^ 1][0][0],
                             Bt + (size_t)n0 * K + (k0 + KB2), KB2, BN, (unsigned)K);
        }

        #pragma unroll
        for (int s = 0; s < 2; ++s) {          // two K=32 sub-steps per stage
            FragA fa;
            #pragma unroll
            for (int v = 0; v < 8; ++v) {
                int kk = 16 * s + ((v >> 2) << 3) + (h << 2) + (v & 3);
                fa.u[v] = As[buf][mw + r][kk];
            }
            #pragma unroll
            for (int nf = 0; nf < 4; ++nf) {
                FragA fb;
                #pragma unroll
                for (int v = 0; v < 8; ++v)
                    fb.u[v] = Bs[buf][nf * 16 + r][16 * s + ((v >> 2) << 3) + (h << 2) + (v & 3)];
                acc[nf] = __builtin_amdgcn_wmma_f32_16x16x32_f16(
                    false, fa.h, false, fb.h, (short)0, acc[nf], false, false);
            }
        }

        __syncthreads();                       // everyone done reading buf
        if (more && wid == 0)
            __builtin_amdgcn_s_wait_tensorcnt(0);  // next buffer landed
        __syncthreads();                       // publish to all waves
        buf ^= 1;
    }

    // fused epilogue: column sum of squares
    #pragma unroll
    for (int nf = 0; nf < 4; ++nf) {
        float s = 0.0f;
        #pragma unroll
        for (int v = 0; v < 8; ++v) s += acc[nf][v] * acc[nf][v];
        s += __shfl_xor(s, 16, 32);            // combine the two half-waves
        if (h == 0) atomicAdd(&red[nf * 16 + (lane & 15)], s); // ds_add_f32
    }
    __syncthreads();
    for (int n = tid; n < BN; n += 256)
        atomicAdd(&S[n0 + n], red[n]);         // global_atomic_add_f32
}

// ---------------------------------------------------------------------------
// Small helper kernels
// ---------------------------------------------------------------------------
__global__ void transpose_kernel(const float* __restrict__ in,
                                 float* __restrict__ out, int n)
{
    __shared__ float t[32][33];
    int x = blockIdx.x * 32 + threadIdx.x;
    int y = blockIdx.y * 32 + threadIdx.y;
    #pragma unroll
    for (int j = 0; j < 32; j += 8)
        t[threadIdx.y + j][threadIdx.x] = in[(size_t)(y + j) * n + x];
    __syncthreads();
    x = blockIdx.y * 32 + threadIdx.x;
    y = blockIdx.x * 32 + threadIdx.y;
    #pragma unroll
    for (int j = 0; j < 32; j += 8)
        out[(size_t)(y + j) * n + x] = t[threadIdx.x][threadIdx.y + j];
}

// Bt[j][i] = (f16)|in[i][j]| : fused abs + transpose + f16 convert
__global__ void abs_transpose_f16_kernel(const float* __restrict__ in,
                                         f16* __restrict__ out, int n)
{
    __shared__ float t[32][33];
    int x = blockIdx.x * 32 + threadIdx.x;
    int y = blockIdx.y * 32 + threadIdx.y;
    #pragma unroll
    for (int j = 0; j < 32; j += 8)
        t[threadIdx.y + j][threadIdx.x] = fabsf(in[(size_t)(y + j) * n + x]);
    __syncthreads();
    x = blockIdx.y * 32 + threadIdx.x;
    y = blockIdx.x * 32 + threadIdx.y;
    #pragma unroll
    for (int j = 0; j < 32; j += 8)
        out[(size_t)(y + j) * n + x] = (f16)t[threadIdx.x][threadIdx.y + j];
}

__global__ void convert_f16_kernel(const float* __restrict__ in,
                                   f16* __restrict__ out, int n) {
    int i = blockIdx.x * blockDim.x + threadIdx.x;
    if (i < n) out[i] = (f16)in[i];
}

__global__ void identity_kernel(float* Q, int n) {
    int i = blockIdx.x * blockDim.x + threadIdx.x;
    if (i < n * n) Q[i] = ((i / n) == (i % n)) ? 1.0f : 0.0f;
}

__global__ void zero_kernel(float* p, int n) {
    int i = blockIdx.x * blockDim.x + threadIdx.x;
    if (i < n) p[i] = 0.0f;
}

// Left-looking Cholesky, single workgroup (small vs GEMM cost).
__global__ __launch_bounds__(1024) void chol_kernel(
    const float* __restrict__ G, float* __restrict__ L, int n)
{
    int tid = threadIdx.x;
    for (int i = tid; i < n * n; i += 1024) {
        int r = i / n, c = i % n;
        L[i] = (r >= c) ? G[i] : 0.0f;
    }
    __syncthreads();
    for (int j = 0; j < n; ++j) {
        float s = 0.0f;
        if (tid >= j) {
            s = L[(size_t)tid * n + j];
            for (int p = 0; p < j; ++p)
                s -= L[(size_t)tid * n + p] * L[(size_t)j * n + p];
            if (tid == j) L[(size_t)j * n + j] = sqrtf(fmaxf(s, 1e-20f));
        }
        __syncthreads();
        if (tid > j) L[(size_t)tid * n + j] = s / L[(size_t)j * n + j];
        __syncthreads();
    }
}

// Row-parallel triangular solve: Q = A * R^{-1} with G = R^T R, R = L^T.
__global__ void trsm_rows_kernel(const float* __restrict__ A,
                                 const float* __restrict__ L,
                                 float* __restrict__ Q, int n)
{
    int r = blockIdx.x * blockDim.x + threadIdx.x;
    if (r >= n) return;
    for (int j = 0; j < n; ++j) {
        float s = A[(size_t)r * n + j];
        for (int p = 0; p < j; ++p)
            s -= Q[(size_t)r * n + p] * L[(size_t)j * n + p];
        Q[(size_t)r * n + j] = s / L[(size_t)j * n + j];
    }
}

// Eigen-frequencies from the (approximate) quasi-triangular Schur form.
__global__ void eigfreq_kernel(const float* __restrict__ T,
                               float* __restrict__ eigfs, int n)
{
    int i = blockIdx.x * blockDim.x + threadIdx.x;
    if (i >= n) return;
    const float tol = 1e-5f;
    float sub_prev = (i > 0)     ? T[(size_t)i * n + (i - 1)] : 0.0f;
    float sub_next = (i < n - 1) ? T[(size_t)(i + 1) * n + i] : 0.0f;
    int r0 = -1;
    if (fabsf(sub_prev) > tol && fabsf(sub_prev) >= fabsf(sub_next)) r0 = i - 1;
    else if (fabsf(sub_next) > tol) r0 = i;

    float fs;
    if (r0 < 0) {
        float lam = T[(size_t)i * n + i];
        fs = (lam < 0.0f) ? 1.0f : 0.0f;
    } else {
        float a = T[(size_t)r0 * n + r0];
        float b = T[(size_t)r0 * n + r0 + 1];
        float c = T[(size_t)(r0 + 1) * n + r0];
        float d = T[(size_t)(r0 + 1) * n + r0 + 1];
        float m = 0.5f * (a + d);
        float disc = 0.25f * (a - d) * (a - d) + b * c;
        if (disc < 0.0f) {
            float im = sqrtf(-disc);
            fs = fabsf(atan2f(im, m)) / (float)M_PI;
        } else {
            float lam = (i == r0) ? (m + sqrtf(disc)) : (m - sqrtf(disc));
            fs = (lam < 0.0f) ? 1.0f : 0.0f;
        }
    }
    eigfs[i] = fs;
}

__global__ void interp_kernel(const float* __restrict__ eigfs,
                              const float* __restrict__ psd,
                              float* __restrict__ target, int nd, int nf)
{
    int i = blockIdx.x * blockDim.x + threadIdx.x;
    if (i >= nd) return;
    float fs = eigfs[i];
    float scale = (float)(nf - 1);
    int idx = (int)floorf(fs * scale + 0.5f);
    idx = min(max(idx, 0), nf - 1);
    float t;
    if (idx == 0)           t = psd[0];
    else if (idx == nf - 1) t = psd[nf - 1];
    else {
        float fi = (float)idx / scale;
        float w1 = (fi - fs) * scale;
        t = w1 * psd[idx - 1] + (1.0f - w1) * psd[idx];
    }
    target[i] = t;
}

__global__ __launch_bounds__(1024) void loss_kernel(
    const float* __restrict__ S, const float* __restrict__ target,
    const float* __restrict__ eigfs, float* __restrict__ out, int n)
{
    __shared__ float buf[1024];
    int i = threadIdx.x;
    float c = S[i];        // 1/B mean factor cancels in the normalization
    float t = target[i];

    buf[i] = c; __syncthreads();
    for (int s = 512; s > 0; s >>= 1) { if (i < s) buf[i] += buf[i + s]; __syncthreads(); }
    float sc = buf[0]; __syncthreads();

    buf[i] = t; __syncthreads();
    for (int s = 512; s > 0; s >>= 1) { if (i < s) buf[i] += buf[i + s]; __syncthreads(); }
    float st = buf[0]; __syncthreads();

    float err = (eigfs[i] == 0.0f) ? 0.0f : (t / st - c / sc);
    buf[i] = err * err; __syncthreads();
    for (int s = 512; s > 0; s >>= 1) { if (i < s) buf[i] += buf[i + s]; __syncthreads(); }
    if (i == 0) out[0] = buf[0] / (float)n * SCALE;
}

// ---------------------------------------------------------------------------
// Launch: QR-algorithm sweeps (WMMA GEMMs) -> eig freq + |V|~|Qtot| ->
// TDM-staged f16 WMMA GEMM column-sum-of-squares -> interp -> loss.
// ---------------------------------------------------------------------------
extern "C" void kernel_launch(void* const* d_in, const int* in_sizes, int n_in,
                              void* d_out, int out_size, void* d_ws, size_t ws_size,
                              hipStream_t stream) {
    const float* dynamics = (const float*)d_in[0];   // D x D
    const float* zk       = (const float*)d_in[1];   // B x D
    const float* psd      = (const float*)d_in[2];   // F
    const int F = in_sizes[2];
    float* out = (float*)d_out;
    (void)n_in; (void)out_size; (void)ws_size;

    const size_t MAT = (size_t)DDIM * DDIM;
    const size_t ZKN = (size_t)NSAMP * DDIM;
    float* Acur = (float*)d_ws;
    float* At   = Acur + MAT;
    float* G    = At   + MAT;
    float* Lm   = G    + MAT;
    float* Q    = Lm   + MAT;
    float* Qa   = Q    + MAT;
    float* Qb   = Qa   + MAT;
    float* Svec = Qb   + MAT;            // [D]
    float* eigfs  = Svec + DDIM;         // [D]
    float* target = eigfs + DDIM;        // [D]
    f16* Ah = (f16*)(target + DDIM);     // zk in f16        [B x D]
    f16* Bt = Ah + ZKN;                  // |V|^T in f16     [D x D]

    dim3 blk256(256), blk1024(1024);
    dim3 gTr(DDIM / 32, DDIM / 32), bTr(32, 8);
    dim3 gGemmDD(DDIM / BN, DDIM / BM);      // (16, 8)
    dim3 gGemmBD(DDIM / BN, NSAMP / BM);     // (16, 64)

    (void)hipMemcpyAsync(Acur, dynamics, MAT * sizeof(float),
                         hipMemcpyDeviceToDevice, stream);
    identity_kernel<<<(int)((MAT + 255) / 256), blk256, 0, stream>>>(Qa, DDIM);
    convert_f16_kernel<<<(int)((ZKN + 255) / 256), blk256, 0, stream>>>(zk, Ah, (int)ZKN);

    float* Qcur = Qa; float* Qnxt = Qb;
    for (int t = 0; t < QR_ITERS; ++t) {
        transpose_kernel<<<gTr, bTr, 0, stream>>>(Acur, At, DDIM);
        gemm_wmma_kernel<<<gGemmDD, blk256, 0, stream>>>(At, Acur, G, DDIM, DDIM, DDIM);
        chol_kernel<<<1, blk1024, 0, stream>>>(G, Lm, DDIM);
        trsm_rows_kernel<<<DDIM / 256, blk256, 0, stream>>>(Acur, Lm, Q, DDIM);
        transpose_kernel<<<gTr, bTr, 0, stream>>>(Lm, At, DDIM);        // At = R
        gemm_wmma_kernel<<<gGemmDD, blk256, 0, stream>>>(At, Q, Acur, DDIM, DDIM, DDIM);
        gemm_wmma_kernel<<<gGemmDD, blk256, 0, stream>>>(Qcur, Q, Qnxt, DDIM, DDIM, DDIM);
        float* tmp = Qcur; Qcur = Qnxt; Qnxt = tmp;
    }

    // |V|^T ~ |Schur vectors|^T in f16 (both GEMM operands contiguous along K)
    abs_transpose_f16_kernel<<<gTr, bTr, 0, stream>>>(Qcur, Bt, DDIM);

    // eigen_coeffs (up to global normalization): S[j] = sum_i (zk@|V|)[i,j]^2
    zero_kernel<<<(DDIM + 255) / 256, blk256, 0, stream>>>(Svec, DDIM);
    gemm16_colsumsq_tdm_kernel<<<gGemmBD, blk256, 0, stream>>>(
        Ah, Bt, Svec, NSAMP, DDIM, DDIM);

    eigfreq_kernel<<<DDIM / 256, blk256, 0, stream>>>(Acur, eigfs, DDIM);
    interp_kernel<<<DDIM / 256, blk256, 0, stream>>>(eigfs, psd, target, DDIM, F);
    loss_kernel<<<1, blk1024, 0, stream>>>(Svec, target, eigfs, out, DDIM);
}
